// GGMLLayer_68796786147867
// MI455X (gfx1250) — matrix operand order repaired and checked by reading, not compiled, not run on hardware
//
#include <hip/hip_runtime.h>
#include <hip/hip_bf16.h>

// ---- problem constants ----
#define O_DIM   8192
#define I_DIM   8192
#define TOKENS  512
#define NBLKS   256            // Q4_0 blocks per output row (I/32)

// ---- tiling ----
#define BM 128
#define BN 128
#define LDH 40                 // LDS row pitch in halves (80B)

typedef _Float16 half2_t __attribute__((ext_vector_type(2)));
typedef _Float16 half8_t __attribute__((ext_vector_type(8)));
typedef _Float16 v16h    __attribute__((ext_vector_type(16)));
typedef float    v8f     __attribute__((ext_vector_type(8)));
typedef unsigned int u32x4 __attribute__((ext_vector_type(4)));
typedef int          i32x8 __attribute__((ext_vector_type(8)));
typedef int          i32x4 __attribute__((ext_vector_type(4)));

// 1 = TDM tensor_load_to_lds for A tile (6-arg clang-23 builtin)
// 0 = per-thread global_load_async_to_lds_b128
#define USE_TDM 1

__device__ __forceinline__ void async_lds_b128(unsigned lds_off, const void* g) {
    asm volatile("global_load_async_to_lds_b128 %0, %1, off"
                 :: "v"(lds_off), "v"((unsigned long long)(uintptr_t)g)
                 : "memory");
}
__device__ __forceinline__ void wait_async0() {
    asm volatile("s_wait_asynccnt 0x0" ::: "memory");
}

#if USE_TDM
// TDM: load 128x32 f16 tile (row-major, tensor stride I_DIM) into LDS with
// hardware row padding 32->40 halves (pad_interval=16 DWORDs, pad_amount=4 DWORDs).
__device__ __forceinline__ void tdm_load_x_tile(const _Float16* gsrc, unsigned lds_off) {
    unsigned long long ga = (unsigned long long)(uintptr_t)gsrc;
    u32x4 g0;
    g0[0] = 1u;                                   // count=1, user descriptor
    g0[1] = lds_off;                              // lds_addr (bytes)
    g0[2] = (unsigned)(ga & 0xFFFFFFFFu);         // global_addr[31:0]
    g0[3] = (unsigned)((ga >> 32) & 0x01FFFFFFu)  // global_addr[56:32]
          | 0x80000000u;                          // type=2 ("image")
    i32x8 g1;
    g1[0] = (1 << 16)                             // data_size: 2 bytes
          | (1 << 20)                             // pad_enable
          | (3 << 22)                             // pad_interval: 16 DWORDs
          | (3 << 25);                            // pad_amount: 4 DWORDs
    g1[1] = (int)((I_DIM & 0xFFFFu) << 16);       // tensor_dim0[15:0]
    g1[2] = (int)(((unsigned)I_DIM >> 16)         // tensor_dim0[31:16]
          | ((TOKENS & 0xFFFFu) << 16));          // tensor_dim1[15:0]
    g1[3] = (int)(((unsigned)TOKENS >> 16)        // tensor_dim1[31:16]
          | (32u << 16));                         // tile_dim0 = 32
    g1[4] = 128;                                  // tile_dim1 = 128 (tile_dim2=0)
    g1[5] = I_DIM;                                // tensor_dim0_stride[31:0]
    g1[6] = 0;
    g1[7] = 0;
    i32x4 z4 = {0, 0, 0, 0};                      // groups 2/3 unused (2D tensor)
    i32x8 z8 = {0, 0, 0, 0, 0, 0, 0, 0};
    __builtin_amdgcn_tensor_load_to_lds(g0, g1, z4, z4, z8, 0);
}
#endif

// f16 magic dequant: byte v (two nibbles) -> {d*(lo-8), d*(hi-8)} via one pk_fma.
__device__ __forceinline__ unsigned dq_pair(unsigned v, unsigned dpk, unsigned cpk) {
    unsigned p = ((v | (v << 12)) & 0x000F000Fu) | 0x64006400u;
    half2_t w = __builtin_bit_cast(half2_t, p) * __builtin_bit_cast(half2_t, dpk)
              + __builtin_bit_cast(half2_t, cpk);
    return __builtin_bit_cast(unsigned, w);
}

__device__ __forceinline__ void dequant8(int4 qa, int4 qb, unsigned dpk, unsigned cpk,
                                         int4& st0, int4& st1) {
    st0.x = (int)dq_pair((unsigned)qa.x, dpk, cpk);
    st0.y = (int)dq_pair((unsigned)qa.y, dpk, cpk);
    st0.z = (int)dq_pair((unsigned)qa.z, dpk, cpk);
    st0.w = (int)dq_pair((unsigned)qa.w, dpk, cpk);
    st1.x = (int)dq_pair((unsigned)qb.x, dpk, cpk);
    st1.y = (int)dq_pair((unsigned)qb.y, dpk, cpk);
    st1.z = (int)dq_pair((unsigned)qb.z, dpk, cpk);
    st1.w = (int)dq_pair((unsigned)qb.w, dpk, cpk);
}

__device__ __forceinline__ void scale_consts(_Float16 dh, unsigned& dpk, unsigned& cpk) {
    _Float16 ch = dh * (_Float16)(-1032.0f);
    unsigned db = (unsigned)__builtin_bit_cast(unsigned short, dh);
    unsigned cb = (unsigned)__builtin_bit_cast(unsigned short, ch);
    dpk = db * 0x10001u;
    cpk = cb * 0x10001u;
}

__device__ __forceinline__ void cvt_interleave(float4 a0, float4 a1, float4 b0, float4 b1,
                                               half8_t& h0, half8_t& h1) {
    h0 = { (_Float16)a0.x, (_Float16)b0.x, (_Float16)a0.y, (_Float16)b0.y,
           (_Float16)a0.z, (_Float16)b0.z, (_Float16)a0.w, (_Float16)b0.w };
    h1 = { (_Float16)a1.x, (_Float16)b1.x, (_Float16)a1.y, (_Float16)b1.y,
           (_Float16)a1.z, (_Float16)b1.z, (_Float16)a1.w, (_Float16)b1.w };
}

// ---- pre-pass: x f32 -> f16 in block-interleaved K order (8 MiB to ws) ----
__global__ __launch_bounds__(256)
void xcvt_kernel(const float* __restrict__ x, _Float16* __restrict__ xf) {
    size_t hb  = (size_t)blockIdx.x * 256 + threadIdx.x;
    size_t blk = hb >> 1;
    int    j0  = (int)(hb & 1) * 8;
    const float* s = x + blk * 32;
    float4 a0 = *(const float4*)(s + j0);
    float4 a1 = *(const float4*)(s + j0 + 4);
    float4 b0 = *(const float4*)(s + 16 + j0);
    float4 b1 = *(const float4*)(s + 16 + j0 + 4);
    half8_t h0, h1;
    cvt_interleave(a0, a1, b0, b1, h0, h1);
    *(half8_t*)(xf + blk * 32 + 2 * j0)     = h0;
    *(half8_t*)(xf + blk * 32 + 2 * j0 + 8) = h1;
}

template<bool XF16>
__global__ __launch_bounds__(256)
void q4_0_gemm_wmma(const float* __restrict__ x,
                    const _Float16* __restrict__ xf,
                    const _Float16* __restrict__ d,
                    const int* __restrict__ qs,
                    const float* __restrict__ bias,
                    float* __restrict__ out)
{
    __shared__ __align__(16) _Float16 As[2][BM * LDH];
    __shared__ __align__(16) _Float16 Bs[2][BN * LDH];

    const int t    = threadIdx.x;
    const int lane = t & 31;
    const int wave = t >> 5;
    const int m_base = (wave >> 2) * 64;
    const int n_base = (wave & 3) * 32;
    const int block_n = blockIdx.x;
    const int block_m = blockIdx.y;

    const int mrow   = lane & 15;
    const int koff_a = (lane >> 4) << 3;
    const int koff_b = (lane >> 4) << 3;
    const int rhalf  = (lane >> 4) << 3;

    v8f acc[4][2];
#pragma unroll
    for (int mi = 0; mi < 4; ++mi)
#pragma unroll
        for (int ni = 0; ni < 2; ++ni)
            acc[mi][ni] = (v8f)(0.0f);

    const int xrow  = t >> 1;
    const int xhalf = t & 1;
    const int xj0   = xhalf * 8;
    const int wn    = t >> 1;
    const int ng    = block_n * BN + wn;
    const size_t qbase = (size_t)ng * NBLKS * 16 + xj0;
    const size_t dbase = (size_t)ng * NBLKS;
    const size_t rowx  = (size_t)(block_m * BM + xrow) * I_DIM;
    const size_t xfbase = rowx + xhalf * 16;
    const unsigned lA = (unsigned)(uintptr_t)&As[0][xrow * LDH + xhalf * 16];
    const unsigned lAbuf0 = (unsigned)(uintptr_t)&As[0][0];
    const unsigned lAbuf1 = (unsigned)(uintptr_t)&As[1][0];
    const unsigned lAstride = lAbuf1 - lAbuf0;
    const _Float16* xtile0 = xf + (size_t)(block_m * BM) * I_DIM;  // tile base, kb=0

    // ---------- prologue: stage kb=0 into buffer 0 ----------
    if (XF16) {
#if USE_TDM
        if (t < 32) tdm_load_x_tile(xtile0, lAbuf0);
#else
        const _Float16* src = xf + xfbase;
        async_lds_b128(lA,      src);
        async_lds_b128(lA + 16, src + 8);
#endif
    } else {
        const float* s = x + rowx + xj0;
        float4 a0 = *(const float4*)(s);
        float4 a1 = *(const float4*)(s + 4);
        float4 b0 = *(const float4*)(s + 16);
        float4 b1 = *(const float4*)(s + 20);
        half8_t h0, h1;
        cvt_interleave(a0, a1, b0, b1, h0, h1);
        *(half8_t*)&As[0][xrow * LDH + 2 * xj0]     = h0;
        *(half8_t*)&As[0][xrow * LDH + 2 * xj0 + 8] = h1;
    }
    {
        const int4* q4 = (const int4*)(qs + qbase);
        int4 qa = q4[0], qb = q4[1];
        unsigned dpk, cpk;
        scale_consts(d[dbase], dpk, cpk);
        int4 st0, st1;
        dequant8(qa, qb, dpk, cpk, st0, st1);
        *(int4*)&Bs[0][wn * LDH + 2 * xj0]     = st0;
        *(int4*)&Bs[0][wn * LDH + 2 * xj0 + 8] = st1;
    }
    if (XF16) {
#if USE_TDM
        if (t < 32) __builtin_amdgcn_s_wait_tensorcnt(0);
#else
        wait_async0();
#endif
    }
    __syncthreads();

    // ---------- pipelined K loop ----------
#pragma unroll 2
    for (int kb = 0; kb < NBLKS; ++kb) {
        const int cur = kb & 1;
        const int nxt = cur ^ 1;
        const bool more = (kb + 1) < NBLKS;

        // (1) kick next A tile into the other buffer
        float4 a0, a1, b0, b1;
        if (more) {
            if (XF16) {
#if USE_TDM
                if (t < 32)
                    tdm_load_x_tile(xtile0 + (size_t)(kb + 1) * 32,
                                    nxt ? lAbuf1 : lAbuf0);
#else
                const _Float16* src = xf + xfbase + (size_t)(kb + 1) * 32;
                async_lds_b128(lA + nxt * lAstride,      src);
                async_lds_b128(lA + nxt * lAstride + 16, src + 8);
#endif
            } else {
                const float* s = x + rowx + (size_t)(kb + 1) * 32 + xj0;
                a0 = *(const float4*)(s);
                a1 = *(const float4*)(s + 4);
                b0 = *(const float4*)(s + 16);
                b1 = *(const float4*)(s + 20);
            }
        }
        // (2) prefetch next qs block + scale into registers
        int4 qa, qb; _Float16 dh = (_Float16)0.0f;
        if (more) {
            const int4* q4 = (const int4*)(qs + qbase + (size_t)(kb + 1) * 16);
            qa = q4[0]; qb = q4[1];
            dh = d[dbase + kb + 1];
        }

        // (3) fragments from current buffer + 8 WMMAs
        v16h afrag[4], bfrag[2];
#pragma unroll
        for (int mi = 0; mi < 4; ++mi) {
            const _Float16* ap = &As[cur][(m_base + mi * 16 + mrow) * LDH + koff_a];
            half8_t alo = *(const half8_t*)ap;
            half8_t ahi = *(const half8_t*)(ap + 16);
            afrag[mi] = __builtin_shufflevector(alo, ahi,
                0,1,2,3,4,5,6,7,8,9,10,11,12,13,14,15);
        }
#pragma unroll
        for (int ni = 0; ni < 2; ++ni) {
            const _Float16* bp = &Bs[cur][(n_base + ni * 16 + mrow) * LDH + koff_b];
            half8_t blo = *(const half8_t*)bp;
            half8_t bhi = *(const half8_t*)(bp + 16);
            bfrag[ni] = __builtin_shufflevector(blo, bhi,
                0,1,2,3,4,5,6,7,8,9,10,11,12,13,14,15);
        }
#pragma unroll
        for (int mi = 0; mi < 4; ++mi)
#pragma unroll
            for (int ni = 0; ni < 2; ++ni)
                acc[mi][ni] = __builtin_amdgcn_wmma_f32_16x16x32_f16(
                    false, afrag[mi], false, bfrag[ni],
                    (short)0, acc[mi][ni], false, false);

        // (4) dequant (pk_fma magic) + commit next tiles
        if (more) {
            unsigned dpk, cpk;
            scale_consts(dh, dpk, cpk);
            int4 st0, st1;
            dequant8(qa, qb, dpk, cpk, st0, st1);
            *(int4*)&Bs[nxt][wn * LDH + 2 * xj0]     = st0;
            *(int4*)&Bs[nxt][wn * LDH + 2 * xj0 + 8] = st1;
            if (!XF16) {
                half8_t h0, h1;
                cvt_interleave(a0, a1, b0, b1, h0, h1);
                *(half8_t*)&As[nxt][xrow * LDH + 2 * xj0]     = h0;
                *(half8_t*)&As[nxt][xrow * LDH + 2 * xj0 + 8] = h1;
            }
        }

        if (XF16) {
#if USE_TDM
            if (t < 32) __builtin_amdgcn_s_wait_tensorcnt(0);
#else
            wait_async0();
#endif
        }
        __syncthreads();
    }

    // ---------- epilogue: bias + f32 store ----------
#pragma unroll
    for (int ni = 0; ni < 2; ++ni) {
        const int col_g = block_n * BN + n_base + ni * 16 + mrow;
        const float bv = bias[col_g];
#pragma unroll
        for (int mi = 0; mi < 4; ++mi) {
            const int row_g = block_m * BM + m_base + mi * 16 + rhalf;
            float* op = out + (size_t)row_g * O_DIM + col_g;
#pragma unroll
            for (int r = 0; r < 8; ++r)
                op[(size_t)r * O_DIM] = acc[mi][ni][r] + bv;
        }
    }
}

extern "C" void kernel_launch(void* const* d_in, const int* in_sizes, int n_in,
                              void* d_out, int out_size, void* d_ws, size_t ws_size,
                              hipStream_t stream) {
    const float*    x    = (const float*)d_in[0];
    const _Float16* d    = (const _Float16*)d_in[1];
    const int*      qs   = (const int*)d_in[2];
    const float*    bias = (const float*)d_in[3];
    float*          out  = (float*)d_out;

    dim3 grid(O_DIM / BN, TOKENS / BM);   // 64 x 4 workgroups
    dim3 block(256);                       // 8 wave32s

    const size_t need = (size_t)TOKENS * I_DIM * sizeof(_Float16);  // 8 MiB
    if (ws_size >= need) {
        _Float16* xf = (_Float16*)d_ws;
        xcvt_kernel<<<(TOKENS * (I_DIM / 16)) / 256, 256, 0, stream>>>(x, xf);
        q4_0_gemm_wmma<true><<<grid, block, 0, stream>>>(x, xf, d, qs, bias, out);
    } else {
        q4_0_gemm_wmma<false><<<grid, block, 0, stream>>>(x, nullptr, d, qs, bias, out);
    }
}
